// MedianGAActivation_506806141066
// MI455X (gfx1250) — matrix operand order repaired and checked by reading, not compiled, not run on hardware
//
#include <hip/hip_runtime.h>

// Problem constants from the reference: B=8, F=64 -> M=512 rows; N=K=8192; D=16, K_taps=2.
#define GM 512
#define GN 8192
#define GK 8192
#define GD 16

typedef _Float16 v8h  __attribute__((ext_vector_type(8)));
typedef _Float16 v16h __attribute__((ext_vector_type(16)));
typedef float    v8f  __attribute__((ext_vector_type(8)));

// ---------------- f32 -> f16 convert ----------------------------------------------------
__global__ void __launch_bounds__(256)
cvt_f16_kernel(const float* __restrict__ in, _Float16* __restrict__ out, int n) {
    int i = blockIdx.x * blockDim.x + threadIdx.x;
    if (i < n) out[i] = (_Float16)in[i];
}

// ---------------- S (KxN, f32, row-major) -> ST (NxK, f16, k-contiguous) ----------------
// 64x64 tiles through LDS: coalesced reads along n, coalesced writes along k.
__global__ void __launch_bounds__(512)
tconv_kernel(const float* __restrict__ S, _Float16* __restrict__ ST) {
    __shared__ _Float16 tile[64][65];
    const int tx = threadIdx.x;   // 0..63
    const int ty = threadIdx.y;   // 0..7
    const int n0 = blockIdx.x << 6;
    const int k0 = blockIdx.y << 6;
    #pragma unroll
    for (int r = ty; r < 64; r += 8)
        tile[r][tx] = (_Float16)S[(size_t)(k0 + r) * GN + (n0 + tx)];
    __syncthreads();
    #pragma unroll
    for (int r = ty; r < 64; r += 8)
        ST[(size_t)(n0 + r) * GK + (k0 + tx)] = tile[tx][r];
}

// A fragment (16x32 f16), lane l: M=l%16; halves 0..7 <- K=c0..c0+7, halves 8..15 <-
// K=16+c0..16+c0+7, c0 = 8*(l>=16). Two 16B loads.
__device__ inline v16h load_a_frag(const _Float16* __restrict__ row, int k, int aoff) {
    v8h lo = *(const v8h*)(row + k + aoff);
    v8h hi = *(const v8h*)(row + k + 16 + aoff);
    v16h a;
    #pragma unroll
    for (int i = 0; i < 8; ++i) { a[i] = lo[i]; a[i + 8] = hi[i]; }
    return a;
}

// ---------------- WMMA GEMM: C[M,N] f32 = A[M,K] f16 (row-major) * BT[N,K] f16 ----------
// Wave32. Each wave computes a 32(M) x 64(N) C tile: 2 A frags x 4 B frags -> 8 WMMAs
// per 12 b128 loads per k-step, software-pipelined (next k-step's fragments are in
// flight while the current 8 WMMAs execute).
// Block = 8 waves arranged 2(M) x 4(N) -> 64 x 256 of C. Grid = (N/256, M/64).
__global__ void __launch_bounds__(256)
gemm_wmma_kernel(const _Float16* __restrict__ A, const _Float16* __restrict__ BT,
                 float* __restrict__ C) {
    const int lane = threadIdx.x & 31;
    const int wave = threadIdx.x >> 5;
    const int l16  = lane & 15;
    const int hi   = lane >> 4;                       // 0 or 1
    const int wm   = wave & 1;                        // 2 waves along M
    const int wn   = wave >> 1;                       // 4 waves along N

    const int mbase = (blockIdx.y << 6) + (wm << 5);
    const int nbase = (blockIdx.x << 8) + (wn << 6);

    const _Float16* ar0 = A + (size_t)(mbase + l16) * GK;
    const _Float16* ar1 = A + (size_t)(mbase + 16 + l16) * GK;
    const int aoff = hi << 3;                         // 0 or 8
    // B fragment (32x16 f16), lane l: N=l%16, same K order as A -> two 16B loads.
    const _Float16* b0 = BT + (size_t)(nbase +  0 + l16) * GK;
    const _Float16* b1 = BT + (size_t)(nbase + 16 + l16) * GK;
    const _Float16* b2 = BT + (size_t)(nbase + 32 + l16) * GK;
    const _Float16* b3 = BT + (size_t)(nbase + 48 + l16) * GK;

    v8f acc[2][4];
    #pragma unroll
    for (int i = 0; i < 2; ++i)
        #pragma unroll
        for (int t = 0; t < 4; ++t) acc[i][t] = (v8f){};

    // Pipeline prologue: fragments for k = 0.
    v16h a0 = load_a_frag(ar0, 0, aoff);
    v16h a1 = load_a_frag(ar1, 0, aoff);
    v16h vb0 = load_a_frag(b0, 0, aoff);
    v16h vb1 = load_a_frag(b1, 0, aoff);
    v16h vb2 = load_a_frag(b2, 0, aoff);
    v16h vb3 = load_a_frag(b3, 0, aoff);

    #pragma unroll 2
    for (int k = 0; k < GK - 32; k += 32) {
        const int kn = k + 32;
        // Issue next k-step's loads first so they overlap the 8 WMMAs below.
        v16h na0 = load_a_frag(ar0, kn, aoff);
        v16h na1 = load_a_frag(ar1, kn, aoff);
        v16h nb0 = load_a_frag(b0, kn, aoff);
        v16h nb1 = load_a_frag(b1, kn, aoff);
        v16h nb2 = load_a_frag(b2, kn, aoff);
        v16h nb3 = load_a_frag(b3, kn, aoff);

        acc[0][0] = __builtin_amdgcn_wmma_f32_16x16x32_f16(false, a0, false, vb0, (short)0, acc[0][0], false, false);
        acc[0][1] = __builtin_amdgcn_wmma_f32_16x16x32_f16(false, a0, false, vb1, (short)0, acc[0][1], false, false);
        acc[0][2] = __builtin_amdgcn_wmma_f32_16x16x32_f16(false, a0, false, vb2, (short)0, acc[0][2], false, false);
        acc[0][3] = __builtin_amdgcn_wmma_f32_16x16x32_f16(false, a0, false, vb3, (short)0, acc[0][3], false, false);
        acc[1][0] = __builtin_amdgcn_wmma_f32_16x16x32_f16(false, a1, false, vb0, (short)0, acc[1][0], false, false);
        acc[1][1] = __builtin_amdgcn_wmma_f32_16x16x32_f16(false, a1, false, vb1, (short)0, acc[1][1], false, false);
        acc[1][2] = __builtin_amdgcn_wmma_f32_16x16x32_f16(false, a1, false, vb2, (short)0, acc[1][2], false, false);
        acc[1][3] = __builtin_amdgcn_wmma_f32_16x16x32_f16(false, a1, false, vb3, (short)0, acc[1][3], false, false);

        a0 = na0; a1 = na1; vb0 = nb0; vb1 = nb1; vb2 = nb2; vb3 = nb3;
    }
    // Epilogue: last k-step.
    acc[0][0] = __builtin_amdgcn_wmma_f32_16x16x32_f16(false, a0, false, vb0, (short)0, acc[0][0], false, false);
    acc[0][1] = __builtin_amdgcn_wmma_f32_16x16x32_f16(false, a0, false, vb1, (short)0, acc[0][1], false, false);
    acc[0][2] = __builtin_amdgcn_wmma_f32_16x16x32_f16(false, a0, false, vb2, (short)0, acc[0][2], false, false);
    acc[0][3] = __builtin_amdgcn_wmma_f32_16x16x32_f16(false, a0, false, vb3, (short)0, acc[0][3], false, false);
    acc[1][0] = __builtin_amdgcn_wmma_f32_16x16x32_f16(false, a1, false, vb0, (short)0, acc[1][0], false, false);
    acc[1][1] = __builtin_amdgcn_wmma_f32_16x16x32_f16(false, a1, false, vb1, (short)0, acc[1][1], false, false);
    acc[1][2] = __builtin_amdgcn_wmma_f32_16x16x32_f16(false, a1, false, vb2, (short)0, acc[1][2], false, false);
    acc[1][3] = __builtin_amdgcn_wmma_f32_16x16x32_f16(false, a1, false, vb3, (short)0, acc[1][3], false, false);

    // C/D layout: VGPR r, lane l -> M = r + 8*(l>=16), N = l%16.
    #pragma unroll
    for (int i = 0; i < 2; ++i) {
        float* crow = C + (size_t)(mbase + (i << 4) + (hi << 3)) * GN + nbase + l16;
        #pragma unroll
        for (int r = 0; r < 8; ++r) {
            crow[(size_t)r * GN +  0] = acc[i][0][r];
            crow[(size_t)r * GN + 16] = acc[i][1][r];
            crow[(size_t)r * GN + 32] = acc[i][2][r];
            crow[(size_t)r * GN + 48] = acc[i][3][r];
        }
    }
}

// ---------------- median-of-16 gather (+ optional fused relu/weight combine) ------------
// Fully-unrolled bitonic sort network -> all 16 values stay in registers; take rank 7
// (jnp.sort(...)[..., (D-1)//2]).
__global__ void __launch_bounds__(256)
median_kernel(const float* __restrict__ cur, const int* __restrict__ nbr,
              const float* __restrict__ xin, const float* __restrict__ t1,
              const float* __restrict__ w, float* __restrict__ outp, int fuse) {
    int i = blockIdx.x * blockDim.x + threadIdx.x;
    if (i >= GM * GN) return;
    const int n = i & (GN - 1);
    const int m = i >> 13;                      // log2(GN)
    const float* row = cur + (size_t)m * GN;

    float g[GD];
    #pragma unroll
    for (int d = 0; d < GD; ++d) {
        int idx = nbr[(size_t)n * GD + d];
        g[d] = row[idx];
    }

    #pragma unroll
    for (int kk = 2; kk <= GD; kk <<= 1) {
        #pragma unroll
        for (int j = kk >> 1; j > 0; j >>= 1) {
            #pragma unroll
            for (int t = 0; t < GD; ++t) {
                int l = t ^ j;
                if (l > t) {
                    float a = g[t], b = g[l];
                    bool up = ((t & kk) == 0);
                    bool sw = up ? (a > b) : (a < b);
                    g[t] = sw ? b : a;
                    g[l] = sw ? a : b;
                }
            }
        }
    }
    float med = g[7];

    if (fuse) {
        float xv = xin[i];
        float r  = xv > 0.0f ? xv : 0.0f;
        outp[i] = w[0] * r + w[1] * t1[i] + w[2] * med;
    } else {
        outp[i] = med;
    }
}

extern "C" void kernel_launch(void* const* d_in, const int* in_sizes, int n_in,
                              void* d_out, int out_size, void* d_ws, size_t ws_size,
                              hipStream_t stream) {
    const float* x  = (const float*)d_in[0];   // (8,64,8192) f32
    const float* S  = (const float*)d_in[1];   // (1,8192,8192) f32
    const float* w  = (const float*)d_in[2];   // (1,3) f32
    const int*   nb = (const int*)d_in[3];     // (8192,16) integer
    float* out = (float*)d_out;                // (8,64,8192) f32

    char* ws = (char*)d_ws;
    _Float16* ST16 = (_Float16*)ws;                                   // 128 MB, L2-resident
    _Float16* A16  = (_Float16*)(ws + (size_t)GN * GK * 2);           // 8 MB (x16, then cur1_16)
    float*    cur  = (float*)(ws + (size_t)GN * GK * 2
                                 + (size_t)GM * GK * 2);              // 16 MB (cur1, then cur2)
    float*    t1   = (float*)((char*)cur + (size_t)GM * GN * 4);      // 16 MB

    const int elems = GM * GK;                                        // 4,194,304
    dim3 gemm_grid(GN / 256, GM / 64);

    // 1) x -> f16
    cvt_f16_kernel<<<(elems + 255) / 256, 256, 0, stream>>>(x, A16, elems);
    // 2) S -> transposed f16 (fits in 192 MB L2; serves both GEMMs)
    tconv_kernel<<<dim3(GN / 64, GK / 64), dim3(64, 8), 0, stream>>>(S, ST16);
    // 3) cur1 = x @ S
    gemm_wmma_kernel<<<gemm_grid, 256, 0, stream>>>(A16, ST16, cur);
    // 4) t1 = median(cur1[neigh])
    median_kernel<<<(elems + 255) / 256, 256, 0, stream>>>(cur, nb, nullptr, nullptr, nullptr, t1, 0);
    // 5) cur1 -> f16 (reuse A16 slot)
    cvt_f16_kernel<<<(elems + 255) / 256, 256, 0, stream>>>(cur, A16, elems);
    // 6) cur2 = cur1 @ S (overwrites cur buffer)
    gemm_wmma_kernel<<<gemm_grid, 256, 0, stream>>>(A16, ST16, cur);
    // 7) out = w0*relu(x) + w1*t1 + w2*median(cur2[neigh])
    median_kernel<<<(elems + 255) / 256, 256, 0, stream>>>(cur, nb, x, t1, w, out, 1);
}
